// GRU_46926812676646
// MI455X (gfx1250) — compile-verified, matrix operand order checked by reading, not run on hardware
//
#include <hip/hip_runtime.h>
#include <hip/hip_bf16.h>
#include <stdint.h>

// ---------------------------------------------------------------- types ----
typedef __attribute__((ext_vector_type(16))) __bf16  v16bf;
typedef __attribute__((ext_vector_type(8)))  float   v8f;
typedef __attribute__((ext_vector_type(4)))  float   f32x4;
typedef __attribute__((ext_vector_type(4)))  unsigned int u32x4;
typedef __attribute__((ext_vector_type(8)))  int     i32x8;
typedef __attribute__((ext_vector_type(4)))  int     i32x4;

static constexpr int N_B = 32;     // batch
static constexpr int T_S = 512;    // time steps
static constexpr int D_I = 512;    // input dim
static constexpr int H_D = 1024;   // hidden dim
static constexpr int G3  = 3072;   // 3*H

// ------------------------------------------------------ WMMA helpers ------
__device__ inline v8f wmma_bf16(v16bf a, v16bf b, v8f c) {
  // D = A(16x32 bf16) * B(32x16 bf16) + C(16x16 f32)
  return __builtin_amdgcn_wmma_f32_16x16x32_bf16(
      /*neg_a=*/false, a, /*neg_b=*/false, b,
      /*c_mod=*/(short)0, c, /*reuse_a=*/false, /*reuse_b=*/false);
}

// A fragment (ISA 7.12.2, 16-bit A 16x32): lane m=lane&15, kbase=k0+8*(lane>>4)
// holds K = kbase..kbase+7 and kbase+16..kbase+23 -> two contiguous 16B runs.
__device__ inline v16bf load_a(const __bf16* row, int kbase) {
  union U { v16bf v; f32x4 f[2]; } u;
  u.f[0] = *reinterpret_cast<const f32x4*>(row + kbase);
  u.f[1] = *reinterpret_cast<const f32x4*>(row + kbase + 16);
  return u.v;
}

// B fragment from pre-packed fragment-major storage: 32B contiguous per lane.
__device__ inline v16bf load_b(const __bf16* p) {
  union U { v16bf v; f32x4 f[2]; } u;
  const f32x4* q = reinterpret_cast<const f32x4*>(p);
  u.f[0] = q[0];
  u.f[1] = q[1];
  return u.v;
}

// ------------------------------------------------- fp32 -> bf16 convert ---
__global__ void f32_to_bf16_kernel(const float* __restrict__ src,
                                   __bf16* __restrict__ dst, int n) {
  int i = blockIdx.x * blockDim.x + threadIdx.x;
  if (i < n) dst[i] = (__bf16)src[i];
}

// --------------------------------------------- pack W into B-frag layout ---
// dst[((ntile*KK + kk)*32 + lane)*16 + e] = W[k][col_base + ntile*16 + (lane&15)]
// with k per ISA 16-bit B layout (mirror of A): kbase = kk*32 + 8*(lane>>4),
// element e -> q=e>>1, k = kbase + (q<4 ? 2q : 16+2(q-4)) + (e&1)
__global__ void pack_w_kernel(const float* __restrict__ w, __bf16* __restrict__ dst,
                              int K, int ntiles, int col_base) {
  int idx = blockIdx.x * blockDim.x + threadIdx.x;
  int KK = K >> 5;
  int total = ntiles * KK * 32 * 16;
  if (idx >= total) return;
  int e    = idx & 15;
  int lane = (idx >> 4) & 31;
  int tmp  = idx >> 9;
  int kk   = tmp % KK;
  int nt   = tmp / KK;
  int n     = nt * 16 + (lane & 15);
  int kbase = kk * 32 + ((lane >> 4) << 3);
  int q     = e >> 1;
  int k     = kbase + ((q < 4) ? (2 * q) : (16 + 2 * (q - 4))) + (e & 1);
  dst[idx] = (__bf16)w[(size_t)k * G3 + col_base + n];
}

// ------------------------------------------------- phase 1: gx = x @ w_i ---
// block = 256 thr (8 waves): 32x64 macrotile, 1 WMMA tile per wave.
// A tile (32x512 bf16, 32 KB) staged in LDS via the Tensor Data Mover.
__global__ __launch_bounds__(256) void gemm_gx_kernel(
    const __bf16* __restrict__ xb, const __bf16* __restrict__ wi_p,
    float* __restrict__ gx) {
  __shared__ __attribute__((aligned(16))) __bf16 smA[32 * D_I];  // 32 KB
  const int tid = threadIdx.x;
  const int mblock = blockIdx.x;   // 512
  const int nblock = blockIdx.y;   // 48
  const __bf16* gsrc = xb + (size_t)mblock * 32 * D_I;

#if __has_builtin(__builtin_amdgcn_tensor_load_to_lds)
  if (tid < 32) {  // wave 0 issues the TDM descriptor (wave-level op)
    unsigned long long ga = (unsigned long long)(uintptr_t)gsrc;
    unsigned lds = (unsigned)(uintptr_t)(&smA[0]);
    u32x4 g0 = {0, 0, 0, 0};
    i32x8 g1 = {0, 0, 0, 0, 0, 0, 0, 0};
    i32x4 g2 = {0, 0, 0, 0};
    i32x4 g3 = {0, 0, 0, 0};
    // D# group0: count=1 | lds_addr | global_addr[56:0] | type=2
    g0[0] = 1u;
    g0[1] = lds;
    g0[2] = (unsigned)(ga & 0xFFFFFFFFu);
    g0[3] = (unsigned)((ga >> 32) & 0x1FFFFFFu) | (2u << 30);
    // D# group1: data_size=1(2B); tensor_dim0=512; tensor_dim1=16384;
    //            tile_dim0=512; tile_dim1=32; tensor_dim0_stride=512
    g1[0] = (1 << 16);                                   // data_size, mask=0
    g1[1] = (int)((D_I & 0xFFFF) << 16);                 // dim0[15:0] @ 63:48
    g1[2] = (int)(((N_B * T_S) & 0xFFFF) << 16);         // dim1[15:0] @ 95:80
    g1[3] = (int)(D_I << 16);                            // tile_dim0 @ 127:112
    g1[4] = 32;                                          // tile_dim1 @ 143:128
    g1[5] = D_I;                                         // dim0_stride @ 191:160
#if defined(__clang_major__) && __clang_major__ >= 23
    i32x8 z8 = {0, 0, 0, 0, 0, 0, 0, 0};
    __builtin_amdgcn_tensor_load_to_lds(g0, g1, g2, g3, z8, 0);
#else
    __builtin_amdgcn_tensor_load_to_lds(g0, g1, g2, g3, 0);
#endif
    __builtin_amdgcn_s_wait_tensorcnt(0);
  }
  __syncthreads();
#else
  for (int i = tid; i < (32 * D_I) / 8; i += 256)
    reinterpret_cast<f32x4*>(smA)[i] = reinterpret_cast<const f32x4*>(gsrc)[i];
  __syncthreads();
#endif

  const int wave = tid >> 5, lane = tid & 31;
  const int mt = wave >> 2;                  // 0..1
  const int ntile = nblock * 4 + (wave & 3); // 0..191
  v8f acc = {};
  const __bf16* arow = &smA[((mt * 16) + (lane & 15)) * D_I];
  const __bf16* bptr = wi_p + (size_t)ntile * (D_I / 32) * 512 + (size_t)lane * 16;
#pragma unroll 4
  for (int kk = 0; kk < D_I / 32; ++kk) {
    int kbase = kk * 32 + ((lane >> 4) << 3);
    acc = wmma_bf16(load_a(arow, kbase), load_b(bptr + kk * 512), acc);
  }
  const int col = ntile * 16 + (lane & 15);
  const int rowbase = mblock * 32 + mt * 16 + ((lane >> 4) << 3);
#pragma unroll
  for (int i = 0; i < 8; ++i)
    gx[(size_t)(rowbase + i) * G3 + col] = acc[i];
}

// ------------------------------------------------------------ init --------
__global__ void init_kernel(const float* __restrict__ init_h,
                            float* __restrict__ h_f32, __bf16* __restrict__ h_bf,
                            float* __restrict__ out_tail, unsigned* __restrict__ bar) {
  int i = blockIdx.x * blockDim.x + threadIdx.x;
  if (i == 0) *bar = 0u;
  if (i < N_B * H_D) {
    float v = init_h[i % H_D];   // h_eff(0) == initial_h for every row
    h_f32[i] = v;
    h_bf[i]  = (__bf16)v;
  }
  if (i < H_D) out_tail[i] = init_h[i];
}

// ------------------------------------------------------ grid-wide sync ----
__device__ inline void grid_sync(unsigned* bar, unsigned& target, int use_cluster) {
  __threadfence();
  __syncthreads();
  target += (unsigned)gridDim.x;
  if (use_cluster) {
    // hardware cluster split-barrier (s_barrier_signal/-wait -3); NOP when
    // the dispatch is not clustered (ISA 3.2), in which case host passes 0.
#if __has_builtin(__builtin_amdgcn_s_cluster_barrier)
    __builtin_amdgcn_s_cluster_barrier();
#else
    asm volatile("s_barrier_signal -3\n\ts_barrier_wait -3" ::: "memory");
#endif
  } else {
    if (threadIdx.x == 0) {
      __hip_atomic_fetch_add(bar, 1u, __ATOMIC_RELEASE, __HIP_MEMORY_SCOPE_AGENT);
      while (__hip_atomic_load(bar, __ATOMIC_ACQUIRE, __HIP_MEMORY_SCOPE_AGENT) < target)
        __builtin_amdgcn_s_sleep(2);
    }
    __syncthreads();
  }
}

// ------------------------------------------------- phase 2: GRU scan ------
// Persistent: 32 WGs x 8 waves = 256 waves == 2 (M) x 128 (N) zr tiles.
__global__ __launch_bounds__(256) void gru_scan_kernel(
    const float* __restrict__ gx, const int* __restrict__ reset,
    const float* __restrict__ bias, const float* __restrict__ init_h,
    const __bf16* __restrict__ whzr_p, const __bf16* __restrict__ wha_p,
    float* __restrict__ h_f32, __bf16* __restrict__ h_bf,
    __bf16* __restrict__ rh_bf, float* __restrict__ z_buf,
    float* __restrict__ out0, float* __restrict__ out1,
    unsigned* __restrict__ bar, int use_cluster) {
  // per-wave zr B-fragments for k<512 (t-invariant): 8 waves * 16KB = 128 KB
  __shared__ __attribute__((aligned(16))) __bf16 smB[8 * 16 * 512];
  const int tid = threadIdx.x, lane = tid & 31, wave = tid >> 5;
  const int gw = blockIdx.x * 8 + wave;       // 0..255
  const int mtA = gw >> 7, ntA = gw & 127;    // zr tile (2 x 128)

  {  // stage first half of this wave's zr B tile into LDS
    const f32x4* src = reinterpret_cast<const f32x4*>(whzr_p + (size_t)ntA * 32 * 512);
    f32x4* dst = reinterpret_cast<f32x4*>(smB + wave * 16 * 512);
    for (int i = lane; i < (16 * 512) / 8; i += 32) dst[i] = src[i];
  }
  __syncthreads();

  unsigned target = 0;
  const int rowadd = (lane >> 4) << 3;

  for (int t = 0; t < T_S; ++t) {
    const int cur = t & 1, nxt = cur ^ 1;

    // ---- phase A: zr = sigmoid(gx_zr + h @ W_zr + b_zr) -------------------
    {
      v8f acc = {};
      const __bf16* arow = h_bf + (size_t)cur * N_B * H_D
                         + ((mtA * 16) + (lane & 15)) * H_D;
      const __bf16* bL = smB + wave * 16 * 512 + lane * 16;                 // kk 0..15
      const __bf16* bG = whzr_p + (size_t)ntA * 32 * 512 + lane * 16;       // kk 16..31
#pragma unroll 4
      for (int kk = 0; kk < 16; ++kk)
        acc = wmma_bf16(load_a(arow, kk * 32 + rowadd), load_b(bL + kk * 512), acc);
#pragma unroll 4
      for (int kk = 16; kk < 32; ++kk)
        acc = wmma_bf16(load_a(arow, kk * 32 + rowadd), load_b(bG + kk * 512), acc);

      const int c = ntA * 16 + (lane & 15);      // 0..2047 (z | r)
#pragma unroll
      for (int i = 0; i < 8; ++i) {
        int row = mtA * 16 + rowadd + i;         // batch 0..31
        float g = gx[((size_t)row * T_S + t) * G3 + c] + bias[c] + acc[i];
        float s = 1.0f / (1.0f + __expf(-g));
        if (c < H_D) {
          z_buf[row * H_D + c] = s;
        } else {
          int ch = c - H_D;
          float hv = h_f32[(size_t)cur * N_B * H_D + row * H_D + ch];
          rh_bf[row * H_D + ch] = (__bf16)(s * hv);
        }
      }
    }
    grid_sync(bar, target, use_cluster);

    // ---- phase B: a = tanh(gx_a + (r*h) @ W_a + b_a); h' = (1-z)h + z*a --
    if (gw < 128) {
      const int mt = gw >> 6, ntB = gw & 63;     // a tile (2 x 64)
      v8f acc = {};
      const __bf16* arow = rh_bf + ((mt * 16) + (lane & 15)) * H_D;
      const __bf16* bptr = wha_p + (size_t)ntB * 32 * 512 + lane * 16;
#pragma unroll 4
      for (int kk = 0; kk < 32; ++kk)
        acc = wmma_bf16(load_a(arow, kk * 32 + rowadd), load_b(bptr + kk * 512), acc);

      const int c = ntB * 16 + (lane & 15);      // 0..1023
#pragma unroll
      for (int i = 0; i < 8; ++i) {
        int row = mt * 16 + rowadd + i;
        float ga = gx[((size_t)row * T_S + t) * G3 + 2 * H_D + c]
                 + bias[2 * H_D + c] + acc[i];
        ga = fminf(fmaxf(ga, -15.0f), 15.0f);
        float e2 = __expf(-2.0f * ga);
        float a  = (1.0f - e2) / (1.0f + e2);
        float z  = z_buf[row * H_D + c];
        float hp = h_f32[(size_t)cur * N_B * H_D + row * H_D + c];
        float hn = (1.0f - z) * hp + z * a;
        size_t oidx = ((size_t)row * T_S + t) * H_D + c;
        out0[oidx] = hn;
        out1[oidx] = hn;
        float heff = hn;                          // apply reset for step t+1
        if (t + 1 < T_S && reset[row * T_S + (t + 1)] != 0) heff = init_h[c];
        h_f32[(size_t)nxt * N_B * H_D + row * H_D + c] = heff;
        h_bf [(size_t)nxt * N_B * H_D + row * H_D + c] = (__bf16)heff;
      }
    }
    grid_sync(bar, target, use_cluster);
  }
}

// ------------------------------------------------------------ launch ------
extern "C" void kernel_launch(void* const* d_in, const int* in_sizes, int n_in,
                              void* d_out, int out_size, void* d_ws, size_t ws_size,
                              hipStream_t stream) {
  (void)in_sizes; (void)n_in; (void)out_size; (void)ws_size;
  const float* x   = (const float*)d_in[0];   // (N,T,D)
  const int*   rst = (const int*)  d_in[1];   // (N,T)
  const float* w_i = (const float*)d_in[2];   // (D,3H)
  const float* w_h = (const float*)d_in[3];   // (H,3H)
  const float* b   = (const float*)d_in[4];   // (3H)
  const float* ih  = (const float*)d_in[5];   // (H)
  float* out = (float*)d_out;

  char* ws = (char*)d_ws;
  size_t off = 0;
  auto alloc = [&](size_t bytes) -> char* {
    char* p = ws + off;
    off = (off + bytes + 255) & ~(size_t)255;
    return p;
  };
  float*  gx     = (float*) alloc((size_t)N_B * T_S * G3 * sizeof(float));   // 201 MB
  __bf16* xb     = (__bf16*)alloc((size_t)N_B * T_S * D_I * sizeof(__bf16)); // 16 MB
  __bf16* wi_p   = (__bf16*)alloc((size_t)192 * 16 * 512 * sizeof(__bf16));  // 3 MB
  __bf16* whzr_p = (__bf16*)alloc((size_t)128 * 32 * 512 * sizeof(__bf16));  // 4 MB
  __bf16* wha_p  = (__bf16*)alloc((size_t)64  * 32 * 512 * sizeof(__bf16));  // 2 MB
  float*  h_f32  = (float*) alloc((size_t)2 * N_B * H_D * sizeof(float));
  __bf16* h_bf   = (__bf16*)alloc((size_t)2 * N_B * H_D * sizeof(__bf16));
  __bf16* rh_bf  = (__bf16*)alloc((size_t)N_B * H_D * sizeof(__bf16));
  float*  z_buf  = (float*) alloc((size_t)N_B * H_D * sizeof(float));
  unsigned* bar  = (unsigned*)alloc(256);

  // 1) x -> bf16
  {
    int n = N_B * T_S * D_I;
    f32_to_bf16_kernel<<<(n + 255) / 256, 256, 0, stream>>>(x, xb, n);
  }
  // 2) pack weights into WMMA B-fragment layout
  pack_w_kernel<<<(192 * 16 * 512 + 255) / 256, 256, 0, stream>>>(w_i, wi_p, D_I, 192, 0);
  pack_w_kernel<<<(128 * 32 * 512 + 255) / 256, 256, 0, stream>>>(w_h, whzr_p, H_D, 128, 0);
  pack_w_kernel<<<(64  * 32 * 512 + 255) / 256, 256, 0, stream>>>(w_h, wha_p,  H_D, 64, 2 * H_D);
  // 3) gates_x = x @ w_i (TDM-staged A tiles, WMMA bf16)
  gemm_gx_kernel<<<dim3(512, 48), 256, 0, stream>>>(xb, wi_p, gx);
  // 4) init h buffers, barrier counter, initial_h output copy
  init_kernel<<<(N_B * H_D + 255) / 256, 256, 0, stream>>>(
      ih, h_f32, h_bf, out + (size_t)2 * N_B * T_S * H_D, bar);
  // 5) persistent recurrent scan (use_cluster=0: plain dispatch, atomic sync)
  gru_scan_kernel<<<32, 256, 0, stream>>>(
      gx, rst, b, ih, whzr_p, wha_p, h_f32, h_bf, rh_bf, z_buf,
      out, out + (size_t)N_B * T_S * H_D, bar, /*use_cluster=*/0);
}
